// FutureDecoder_79963701116991
// MI455X (gfx1250) — matrix operand order repaired and checked by reading, not compiled, not run on hardware
//
#include <hip/hip_runtime.h>
#include <hip/hip_bf16.h>

// ---------------------------------------------------------------------------
// Types / helpers for CDNA5 WMMA
// ---------------------------------------------------------------------------
typedef __attribute__((ext_vector_type(16))) __bf16 v16bf;
typedef __attribute__((ext_vector_type(2)))  __bf16 v2bf;
typedef __attribute__((ext_vector_type(8)))  float  v8f;

__device__ __forceinline__ __bf16 f2bf(float f) {
  unsigned short h = (unsigned short)(__builtin_bit_cast(unsigned, f) >> 16);
  return __builtin_bit_cast(__bf16, h);
}
// pack two f32 as adjacent bf16 (lo = first)
__device__ __forceinline__ unsigned pack2(float a, float b) {
  unsigned ua = __builtin_bit_cast(unsigned, a) >> 16;
  unsigned ub = __builtin_bit_cast(unsigned, b) & 0xffff0000u;
  return ua | ub;
}

#define SA 17   // dword stride of an A row in LDS (conflict-free: gcd(17,32)=1)
#define SB 17   // dword stride of a B column in LDS

// ---------------------------------------------------------------------------
// Guard-free GEMM: C[M,N] = act(A[M,K] @ W[K,N] + bias [+ res])
// REQUIRES: M % 128 == 0, N % 64 == 0, K % 32 == 0 (host pads to guarantee).
// Block 256 = 8 waves (4x2); block tile 128x64; wave tile 32x32 = four
// v_wmma_f32_16x16x32_bf16 accumulators (8 LDS dwords gathered per WMMA).
// ---------------------------------------------------------------------------
__global__ __launch_bounds__(256)
void gemm_bf16_wmma(const float* __restrict__ A, const float* __restrict__ W,
                    const float* __restrict__ bias, const float* __restrict__ res,
                    float* __restrict__ C, int M, int N, int K, int relu)
{
  __shared__ unsigned lA[128 * SA];  // [row][k/2] bf16 pairs
  __shared__ unsigned lB[64 * SB];   // [n][k/2]  bf16 pairs (transposed)

  const int t    = threadIdx.x;
  const int lane = t & 31;
  const int wid  = t >> 5;
  const int mblk = blockIdx.y * 128;
  const int nblk = blockIdx.x * 64;
  const int wm   = (wid & 3) * 32;   // wave tile origin inside block
  const int wn   = (wid >> 2) * 32;
  const int hi   = lane >> 4;
  const int l15  = lane & 15;

  // cooperative-load assignments
  const int ar  = t >> 3;   // 0..31 (rows ar + 32*i)
  const int ak4 = t & 7;    // float4 index within 32-wide K slab
  const int bk  = t >> 4;   // 0..15 (k rows bk + 16*i)
  const int bn4 = t & 15;   // float4 index within 64-wide N slab

  v8f c00 = {}, c01 = {}, c10 = {}, c11 = {};

  for (int k0 = 0; k0 < K; k0 += 32) {
    __syncthreads();
    // ---- A tile: rows [mblk, mblk+128), k [k0, k0+32) ----
#pragma unroll
    for (int i = 0; i < 4; ++i) {
      const int r = ar + 32 * i;
      float4 f = *((const float4*)(A + (size_t)(mblk + r) * K + k0) + ak4);
      lA[r * SA + 2 * ak4]     = pack2(f.x, f.y);
      lA[r * SA + 2 * ak4 + 1] = pack2(f.z, f.w);
    }
    // ---- B tile (transposed): k [k0,k0+32), n [nblk, nblk+64) ----
#pragma unroll
    for (int i = 0; i < 2; ++i) {
      const int kk = bk + 16 * i;
      float4 f = *((const float4*)(W + (size_t)(k0 + kk) * N + nblk) + bn4);
      __bf16* lBh = (__bf16*)lB;               // half stride per n = 2*SB
      lBh[(bn4 * 4 + 0) * (2 * SB) + kk] = f2bf(f.x);
      lBh[(bn4 * 4 + 1) * (2 * SB) + kk] = f2bf(f.y);
      lBh[(bn4 * 4 + 2) * (2 * SB) + kk] = f2bf(f.z);
      lBh[(bn4 * 4 + 3) * (2 * SB) + kk] = f2bf(f.w);
    }
    if (k0 + 32 < K)  // prefetch next A slab (global_prefetch_b8)
      __builtin_prefetch(A + (size_t)(mblk + ar) * K + k0 + 32, 0, 1);
    __syncthreads();

    // ---- fragments from LDS (contiguous k = 16*hi + element, same for A and B) ----
    v16bf a0, a1, b0, b1;
#pragma unroll
    for (int v = 0; v < 8; ++v) {
      const int pA = v + hi * 8;                          // A: k=2p,2p+1
      v2bf p0 = __builtin_bit_cast(v2bf, lA[(wm + l15) * SA + pA]);
      v2bf p1 = __builtin_bit_cast(v2bf, lA[(wm + 16 + l15) * SA + pA]);
      a0[2 * v] = p0.x; a0[2 * v + 1] = p0.y;
      a1[2 * v] = p1.x; a1[2 * v + 1] = p1.y;
      const int pB = v + hi * 8;                          // B: k=2p,2p+1
      v2bf q0 = __builtin_bit_cast(v2bf, lB[(wn + l15) * SB + pB]);
      v2bf q1 = __builtin_bit_cast(v2bf, lB[(wn + 16 + l15) * SB + pB]);
      b0[2 * v] = q0.x; b0[2 * v + 1] = q0.y;
      b1[2 * v] = q1.x; b1[2 * v + 1] = q1.y;
    }
    c00 = __builtin_amdgcn_wmma_f32_16x16x32_bf16(false, a0, false, b0,
                                                  (short)0, c00, false, false);
    c01 = __builtin_amdgcn_wmma_f32_16x16x32_bf16(false, a0, false, b1,
                                                  (short)0, c01, false, false);
    c10 = __builtin_amdgcn_wmma_f32_16x16x32_bf16(false, a1, false, b0,
                                                  (short)0, c10, false, false);
    c11 = __builtin_amdgcn_wmma_f32_16x16x32_bf16(false, a1, false, b1,
                                                  (short)0, c11, false, false);
  }

  // ---- epilogue: C/D layout VGPR v -> M = v + hi*8, N = lane&15 ----
  const int n0 = nblk + wn + l15;
  const int n1 = n0 + 16;
  const float bn0 = bias[n0];
  const float bn1 = bias[n1];
#pragma unroll
  for (int v = 0; v < 8; ++v) {
    const int m0 = mblk + wm + v + hi * 8;
    const int m1 = m0 + 16;
    float v00 = c00[v] + bn0, v01 = c01[v] + bn1;
    float v10 = c10[v] + bn0, v11 = c11[v] + bn1;
    if (res) {
      v00 += res[(size_t)m0 * N + n0]; v01 += res[(size_t)m0 * N + n1];
      v10 += res[(size_t)m1 * N + n0]; v11 += res[(size_t)m1 * N + n1];
    }
    if (relu) {
      v00 = fmaxf(v00, 0.0f); v01 = fmaxf(v01, 0.0f);
      v10 = fmaxf(v10, 0.0f); v11 = fmaxf(v11, 0.0f);
    }
    C[(size_t)m0 * N + n0] = v00; C[(size_t)m0 * N + n1] = v01;
    C[(size_t)m1 * N + n0] = v10; C[(size_t)m1 * N + n1] = v11;
  }
}

// ---------------------------------------------------------------------------
// Fused attention: one block per (query-row, head). Block = 64 threads.
// Q,K,V rows = l*B + b, stride 768. Lk <= 64, DH = 64.
// ---------------------------------------------------------------------------
__global__ __launch_bounds__(64)
void attn_kernel(const float* __restrict__ Q, const float* __restrict__ Km,
                 const float* __restrict__ V, const float* __restrict__ amask,
                 float* __restrict__ O, int Lq, int Lk, int Bc, int selfmask)
{
  (void)Lq;
  const int tid = threadIdx.x;
  const int h   = blockIdx.y;
  const int qi  = blockIdx.x;
  const int lq  = qi / Bc;
  const int b   = qi % Bc;

  __shared__ float sl[64];
  __shared__ float red[2];

  if (tid < Lk) {
    const float* qp = Q  + (size_t)qi * 768 + h * 64;
    const float* kp = Km + (size_t)(tid * Bc + b) * 768 + h * 64;
    float acc = 0.0f;
    for (int d = 0; d < 64; ++d) acc += qp[d] * kp[d];
    acc *= 0.125f;
    acc += amask[(lq & 7) * 8 + (tid & 7)];
    if (selfmask && ((tid >> 3) > (lq >> 3))) acc = -__builtin_inff();
    sl[tid] = acc;
  }
  __syncthreads();
  if (tid == 0) {
    float mx = -__builtin_inff();
    for (int k = 0; k < Lk; ++k) mx = fmaxf(mx, sl[k]);
    red[0] = mx;
  }
  __syncthreads();
  if (tid < Lk) sl[tid] = __expf(sl[tid] - red[0]);
  __syncthreads();
  if (tid == 0) {
    float s = 0.0f;
    for (int k = 0; k < Lk; ++k) s += sl[k];
    red[1] = s;
  }
  __syncthreads();
  const float inv = 1.0f / red[1];
  float acc = 0.0f;                         // tid = head dim (0..63)
  for (int k = 0; k < Lk; ++k)
    acc += sl[k] * V[(size_t)(k * Bc + b) * 768 + h * 64 + tid];
  O[(size_t)qi * 768 + h * 64 + tid] = acc * inv;
}

// ---------------------------------------------------------------------------
// LayerNorm in place over D=768 (one block per row)
// ---------------------------------------------------------------------------
__global__ __launch_bounds__(256)
void ln_kernel(float* __restrict__ X, const float* __restrict__ g,
               const float* __restrict__ bta)
{
  const int row = blockIdx.x, tid = threadIdx.x;
  __shared__ float ss[256], sq[256];
  float s = 0.0f, q = 0.0f;
  for (int d = tid; d < 768; d += 256) {
    float v = X[(size_t)row * 768 + d];
    s += v; q += v * v;
  }
  ss[tid] = s; sq[tid] = q;
  __syncthreads();
  for (int off = 128; off; off >>= 1) {
    if (tid < off) { ss[tid] += ss[tid + off]; sq[tid] += sq[tid + off]; }
    __syncthreads();
  }
  const float mu  = ss[0] * (1.0f / 768.0f);
  const float var = sq[0] * (1.0f / 768.0f) - mu * mu;
  const float inv = rsqrtf(var + 1e-5f);
  for (int d = tid; d < 768; d += 256) {
    float v = X[(size_t)row * 768 + d];
    X[(size_t)row * 768 + d] = (v - mu) * inv * g[d] + bta[d];
  }
}

// sinusoidal positional encoding add: token l uses PE[l/8]
__global__ void add_pe(float* __restrict__ H, int Bc)
{
  const int row = blockIdx.x;
  const float pos = (float)((row / Bc) >> 3);
  for (int d = threadIdx.x; d < 768; d += blockDim.x) {
    const int i2 = d & ~1;
    const float dv = __expf((float)i2 * (-9.210340371976184f / 768.0f));
    const float ang = pos * dv;
    H[(size_t)row * 768 + d] += (d & 1) ? __cosf(ang) : __sinf(ang);
  }
}

// zero-padded 2D copy: d[dR,dC] = s[sR,sC] padded with zeros
__global__ void pad2d(const float* __restrict__ s, float* __restrict__ d,
                      int sR, int sC, int dR, int dC)
{
  const int i = blockIdx.x * blockDim.x + threadIdx.x;
  if (i >= dR * dC) return;
  const int r = i / dC, c = i % dC;
  d[i] = (r < sR && c < sC) ? s[(size_t)r * sC + c] : 0.0f;
}

// build initial dec_in_z (padded stride 96): [dec_in[7,a,b,:35], z[a,:32], 0...]
__global__ void build_x0(const float* __restrict__ dec_in,
                         const float* __restrict__ z, float* __restrict__ X)
{
  const int idx = blockIdx.x * blockDim.x + threadIdx.x;
  if (idx >= 8 * 16 * 96) return;
  const int row = idx / 96, f = idx % 96;
  const int a = row / 16, b = row % 16;
  float v = 0.0f;
  if (f < 35)       v = dec_in[((7 * 8 + a) * 16 + b) * 35 + f];
  else if (f < 67)  v = z[a * 32 + (f - 35)];
  X[idx] = v;
}

// append last A frames of seq_out [Lold*16, 64-padded] (+z) at token Lold
__global__ void append_x(const float* __restrict__ seq, const float* __restrict__ z,
                         float* __restrict__ X, int Lold)
{
  const int idx = blockIdx.x * blockDim.x + threadIdx.x;
  if (idx >= 8 * 16 * 96) return;
  const int row = idx / 96, f = idx % 96;
  const int l = row / 16, b = row % 16;
  const size_t drow = (size_t)(Lold + l) * 16 + b;
  float v = 0.0f;
  if (f < 35)       v = seq[((size_t)(Lold - 8 + l) * 16 + b) * 64 + f];
  else if (f < 67)  v = z[l * 32 + (f - 35)];
  X[drow * 96 + f] = v;
}

__global__ void copy_out_k(const float* __restrict__ s, float* __restrict__ d, int n)
{
  const int i = blockIdx.x * blockDim.x + threadIdx.x;
  if (i >= n) return;
  const int row = i / 35, f = i % 35;
  d[i] = s[(size_t)row * 64 + f];
}

// ---------------------------------------------------------------------------
// Host orchestration
// ---------------------------------------------------------------------------
extern "C" void kernel_launch(void* const* d_in, const int* in_sizes, int n_in,
                              void* d_out, int out_size, void* d_ws, size_t ws_size,
                              hipStream_t stream)
{
  (void)in_sizes; (void)n_in; (void)out_size; (void)ws_size;

  const float* dec_in  = (const float*)d_in[0];   // [8,8,16,35]
  const float* context = (const float*)d_in[1];   // [64,16,768] -> [1024,768]
  const float* z       = (const float*)d_in[2];   // [8,32]
  const float* amask   = (const float*)d_in[3];   // [8,8]
  const float* in_b    = (const float*)d_in[4];
  const float* in_w    = (const float*)d_in[5];   // [67,768]
  auto LP = [&](int layer, int leaf) -> const float* {
    return (const float*)d_in[6 + layer * 26 + leaf];
  };
  // leaf ids: 0 b1,1 b2,2 c_bk,3 c_bo,4 c_bq,5 c_bv,6 c_wk,7 c_wo,8 c_wq,9 c_wv,
  // 10 ln1_b,11 ln1_g,12 ln2_b,13 ln2_g,14 ln3_b,15 ln3_g,
  // 16 s_bk,17 s_bo,18 s_bq,19 s_bv,20 s_wk,21 s_wo,22 s_wq,23 s_wv,24 w1,25 w2
  const float* mlp_b0 = (const float*)d_in[110];
  const float* mlp_b1 = (const float*)d_in[111];
  const float* mlp_w0 = (const float*)d_in[112]; // [768,512]
  const float* mlp_w1 = (const float*)d_in[113]; // [512,256]
  const float* out_b  = (const float*)d_in[114]; // [35]
  const float* out_w  = (const float*)d_in[115]; // [256,35]

  float* ws = (float*)d_ws;
  size_t off = 0;
  auto alloc = [&](size_t n) { size_t o = off; off += (n + 63) & ~(size_t)63; return o; };
  float* Xp   = ws + alloc(48 * 16 * 96);    // dec_in_z, K padded 67->96
  float* INW  = ws + alloc(96 * 768);        // in_w zero-padded
  float* OUTW = ws + alloc(256 * 64);        // out_w zero-padded N 35->64
  float* OUTB = ws + alloc(64);
  float* H   = ws + alloc(768 * 768);
  float* Qb  = ws + alloc(768 * 768);
  float* Kb  = ws + alloc(768 * 768);
  float* Vb  = ws + alloc(768 * 768);
  float* T1  = ws + alloc(768 * 768);
  float* FF  = ws + alloc(768 * 3072);
  float* CK[4], *CV[4];
  for (int l = 0; l < 4; ++l) {
    CK[l] = ws + alloc(1024 * 768);
    CV[l] = ws + alloc(1024 * 768);
  }
  float* M0b = ws + alloc(768 * 512);
  float* M1b = ws + alloc(768 * 256);
  float* SEQ = ws + alloc(768 * 64);         // seq_out, N padded 35->64

  auto gemm = [&](const float* A, const float* W, const float* bias,
                  const float* res, float* C, int M, int N, int K, int relu) {
    dim3 grid(N / 64, M / 128);
    gemm_bf16_wmma<<<grid, 256, 0, stream>>>(A, W, bias, res, C, M, N, K, relu);
  };

  // ---- one-time padded weight copies ----
  pad2d<<<(96 * 768 + 255) / 256, 256, 0, stream>>>(in_w, INW, 67, 768, 96, 768);
  pad2d<<<(256 * 64 + 255) / 256, 256, 0, stream>>>(out_w, OUTW, 256, 35, 256, 64);
  pad2d<<<1, 64, 0, stream>>>(out_b, OUTB, 1, 35, 1, 64);

  // ---- step-invariant cross-attention K/V of the context (hoisted) ----
  for (int l = 0; l < 4; ++l) {
    gemm(context, LP(l, 6), LP(l, 2), nullptr, CK[l], 1024, 768, 768, 0); // c_wk
    gemm(context, LP(l, 9), LP(l, 5), nullptr, CV[l], 1024, 768, 768, 0); // c_wv
  }

  build_x0<<<(8 * 16 * 96 + 255) / 256, 256, 0, stream>>>(dec_in, z, Xp);

  // ---- autoregressive rollout, 6 steps ----
  for (int t = 1; t <= 6; ++t) {
    const int Ltok = 8 * t;
    const int M = Ltok * 16;     // multiple of 128

    gemm(Xp, INW, in_b, nullptr, H, M, 768, 96, 0);        // embedding
    add_pe<<<M, 256, 0, stream>>>(H, 16);

    for (int l = 0; l < 4; ++l) {
      // ---- self attention ----
      gemm(H, LP(l, 22), LP(l, 18), nullptr, Qb, M, 768, 768, 0); // s_wq
      gemm(H, LP(l, 20), LP(l, 16), nullptr, Kb, M, 768, 768, 0); // s_wk
      gemm(H, LP(l, 23), LP(l, 19), nullptr, Vb, M, 768, 768, 0); // s_wv
      attn_kernel<<<dim3(M, 12), 64, 0, stream>>>(Qb, Kb, Vb, amask, T1,
                                                  Ltok, Ltok, 16, 1);
      gemm(T1, LP(l, 21), LP(l, 17), H, H, M, 768, 768, 0);       // s_wo + res
      ln_kernel<<<M, 256, 0, stream>>>(H, LP(l, 11), LP(l, 10));  // ln1

      // ---- cross attention (K/V precomputed) ----
      gemm(H, LP(l, 8), LP(l, 4), nullptr, Qb, M, 768, 768, 0);   // c_wq
      attn_kernel<<<dim3(M, 12), 64, 0, stream>>>(Qb, CK[l], CV[l], amask, T1,
                                                  Ltok, 64, 16, 0);
      gemm(T1, LP(l, 7), LP(l, 3), H, H, M, 768, 768, 0);         // c_wo + res
      ln_kernel<<<M, 256, 0, stream>>>(H, LP(l, 13), LP(l, 12));  // ln2

      // ---- FFN ----
      gemm(H, LP(l, 24), LP(l, 0), nullptr, FF, M, 3072, 768, 1); // w1 + relu
      gemm(FF, LP(l, 25), LP(l, 1), H, H, M, 768, 3072, 0);       // w2 + res
      ln_kernel<<<M, 256, 0, stream>>>(H, LP(l, 15), LP(l, 14));  // ln3
    }

    // ---- output MLP ----
    gemm(H,   mlp_w0, mlp_b0, nullptr, M0b, M, 512, 768, 1);
    gemm(M0b, mlp_w1, mlp_b1, nullptr, M1b, M, 256, 512, 1);
    gemm(M1b, OUTW,  OUTB,  nullptr, SEQ, M, 64, 256, 0);

    if (t < 6)
      append_x<<<(8 * 16 * 96 + 255) / 256, 256, 0, stream>>>(SEQ, z, Xp, Ltok);
  }

  copy_out_k<<<(48 * 16 * 35 + 255) / 256, 256, 0, stream>>>(SEQ, (float*)d_out,
                                                             48 * 16 * 35);
}
